// Interactor_31576599560333
// MI455X (gfx1250) — matrix operand order repaired and checked by reading, not compiled, not run on hardware
//
#include <hip/hip_runtime.h>
#include <hip/hip_bf16.h>

#define DEV __device__ __forceinline__

typedef __attribute__((ext_vector_type(16))) _Float16 v16h;
typedef __attribute__((ext_vector_type(8)))  float    v8f;

namespace {

constexpr int EMB  = 256;
constexpr int NB   = 3;
constexpr int NG   = 50;
constexpr int NGP  = 64;            // rbf K padded to multiple of 32
constexpr int G    = 256;
constexpr int ATOMS = 32;
constexpr int N    = G * ATOMS;     // 8192
constexpr int E    = 65536;
constexpr int EPG  = E / G;         // 256 edges per graph
constexpr float LOG2F_ = 0.69314718055994531f;

// shifted softplus via HW transcendentals (v_exp_f32 / v_log_f32):
// log(1+e^x) - log2 == max(x,0) + log(1+e^-|x|) - log2   (stable both tails)
DEV float ssp_f(float x) {
  return fmaxf(x, 0.f) + __logf(1.f + __expf(-fabsf(x))) - LOG2F_;
}

// ---------------- WMMA fragment helpers (wave32, v_wmma_f32_16x16x32_f16) ---
// A (16x32 f16): lane r=lane&15 holds row r. VGPR p holds K pair:
//   lanes 0-15 : p<4 -> K=2p,2p+1   ; p>=4 -> K=16+2(p-4), +1
//   lanes 16-31: same + 8
DEV v16h load_frag_a(const _Float16* A, int ld) {
  const int lane = (int)(threadIdx.x & 31);
  const int r  = lane & 15;
  const int hi = (lane >> 4) & 1;
  v16h a;
#pragma unroll
  for (int p = 0; p < 8; ++p) {
    int k = ((p < 4) ? (2 * p) : (16 + 2 * (p - 4))) + hi * 8;
    a[2 * p]     = A[r * ld + k];
    a[2 * p + 1] = A[r * ld + k + 1];
  }
  return a;
}

// B (32x16 f16, rows = K): lane n=lane&15 is the column. VGPR p holds K pair:
//   lanes 0-15 : p<4 -> K=2p,2p+1   ; p>=4 -> K=16+2(p-4), +1
//   lanes 16-31: same + 8
DEV v16h load_frag_b(const _Float16* B, int ld) {
  const int lane = (int)(threadIdx.x & 31);
  const int n  = lane & 15;
  const int hi = (lane >> 4) & 1;
  v16h b;
#pragma unroll
  for (int p = 0; p < 8; ++p) {
    int k = ((p < 4) ? (2 * p) : (16 + 2 * (p - 4))) + hi * 8;
    b[2 * p]     = B[k * ld + n];
    b[2 * p + 1] = B[(k + 1) * ld + n];
  }
  return b;
}

DEV v8f wmma_f16(v16h a, v16h b, v8f c) {
  // (neg_a, A, neg_b, B, c_mod, C, reuse_a, reuse_b)
  return __builtin_amdgcn_wmma_f32_16x16x32_f16(false, a, false, b,
                                                (short)0, c, false, false);
}

// ---------------------------------------------------------------------------
// Generic tiled GEMM: Out[M,Nout] = act(In[M,K] @ W[K,Nout] + bias)
// In is f32 (staged to f16 in LDS); W is pre-converted f16.
// Block = 256 threads = 8 waves; tile 64 (M) x 64 (N); grid = (Nout/64, M/64).
// Wave w owns M-strip mt = w>>1 and two N subtiles (A-frag reused across 2
// WMMAs per K chunk). ACT: 0 none, 1 relu, 2 shifted-softplus
// ---------------------------------------------------------------------------
template <int ACT>
__global__ __launch_bounds__(256) void gemm_kernel(
    const float* __restrict__ In, int ldin,
    const _Float16* __restrict__ Wt, int ldw,
    const float* __restrict__ bias,
    float* __restrict__ Out, int ldout,
    int K) {
  const int n0 = blockIdx.x * 64;
  const int m0 = blockIdx.y * 64;
  const int t  = (int)threadIdx.x;
  const int w  = t >> 5;
  const int mt  = w >> 1;    // 0..3 : 16-row strip
  const int ntp = w & 1;     // 0..1 : pair of 16-col subtiles

  __shared__ _Float16 At[64 * 40];
  __shared__ _Float16 Bt[32 * 72];

  v8f acc0 = {}, acc1 = {};
  for (int kc = 0; kc < K; kc += 32) {
    {  // stage A chunk (64x32), f32 -> f16
      int r = t >> 2, c0 = (t & 3) * 8;
      const float* src = &In[(size_t)(m0 + r) * ldin + kc + c0];
#pragma unroll
      for (int u = 0; u < 8; ++u) At[r * 40 + c0 + u] = (_Float16)src[u];
    }
    {  // stage B chunk (32x64, f16)
      int r = t >> 3, c0 = (t & 7) * 8;
      const _Float16* src = &Wt[(size_t)(kc + r) * ldw + n0 + c0];
#pragma unroll
      for (int u = 0; u < 8; ++u) Bt[r * 72 + c0 + u] = src[u];
    }
    if (kc + 32 < K)
      __builtin_prefetch(&Wt[(size_t)(kc + 32) * ldw + n0], 0, 1);
    __syncthreads();
    v16h a  = load_frag_a(&At[(mt * 16) * 40], 40);
    v16h b0 = load_frag_b(&Bt[(ntp * 2 + 0) * 16], 72);
    v16h b1 = load_frag_b(&Bt[(ntp * 2 + 1) * 16], 72);
    acc0 = wmma_f16(a, b0, acc0);
    acc1 = wmma_f16(a, b1, acc1);
    __syncthreads();
  }

  const int lane = t & 31;
  const int n  = lane & 15;
  const int mb = (lane >> 4) * 8;
#pragma unroll
  for (int q = 0; q < 2; ++q) {
    const v8f& acc = q ? acc1 : acc0;
    const int gn = n0 + (ntp * 2 + q) * 16 + n;
    const float bv = bias ? bias[gn] : 0.f;
#pragma unroll
    for (int p = 0; p < 8; ++p) {
      float v = acc[p] + bv;
      if (ACT == 1) v = fmaxf(v, 0.f);
      if (ACT == 2) v = ssp_f(v);
      Out[(size_t)(m0 + mt * 16 + mb + p) * ldout + gn] = v;
    }
  }
}

// ---------------------------------------------------------------------------
// Geometry: per-graph pairwise distances -> ccut (f32) and padded RBF (f16)
// ---------------------------------------------------------------------------
__global__ __launch_bounds__(256) void geom_kernel(
    const float* __restrict__ pos, _Float16* __restrict__ rbf,
    float* __restrict__ ccut) {
  const int g = blockIdx.x;
  const int t = (int)threadIdx.x;
  __shared__ float sp[96];
  if (t < 96) sp[t] = pos[(size_t)g * 96 + t];
  __syncthreads();
  const float step  = 10.f / 49.f;
  const float coeff = -0.5f / (step * step);
  for (int p = t; p < 1024; p += 256) {
    int i = p >> 5, j = p & 31;
    float dx = sp[i * 3 + 0] - sp[j * 3 + 0];
    float dy = sp[i * 3 + 1] - sp[j * 3 + 1];
    float dz = sp[i * 3 + 2] - sp[j * 3 + 2];
    float sq = dx * dx + dy * dy + dz * dz;
    float dist = (sq > 0.f) ? sqrtf(sq) : 0.f;
    float adj = ((dist < 10.f) && (i != j)) ? 1.f : 0.f;
    ccut[(size_t)g * 1024 + p] =
        0.5f * (__cosf(dist * 0.31415926535f) + 1.f) * adj;
    size_t base = ((size_t)g * 1024 + p) * NGP;
#pragma unroll 10
    for (int k = 0; k < NG; ++k) {
      float d = dist - (float)k * step;
      rbf[base + k] = (_Float16)__expf(coeff * d * d);
    }
#pragma unroll
    for (int k = NG; k < NGP; ++k) rbf[base + k] = (_Float16)0.f;
  }
}

// ---------------------------------------------------------------------------
// Embedding gather (also seeds residual buffers)
// ---------------------------------------------------------------------------
__global__ void embed_kernel(const int* __restrict__ cls,
                             const float* __restrict__ e2,
                             const float* __restrict__ e3,
                             float* __restrict__ x2d, float* __restrict__ x3d,
                             float* __restrict__ p2, float* __restrict__ p3) {
  const int nid = blockIdx.x;
  const int f   = (int)threadIdx.x;
  const int c   = cls[nid];
  float a = e2[(size_t)c * EMB + f];
  float b = e3[(size_t)c * EMB + f];
  size_t o = (size_t)nid * EMB + f;
  x2d[o] = a; p2[o] = a;
  x3d[o] = b; p3[o] = b;
}

// ---------------------------------------------------------------------------
// GIN aggregation, deterministic (row ownership, no atomics):
//   hin = (1+eps)*x2d + segsum_row( relu(x2d[col] + bond[attr]) )
// One block per graph; node features + bond table staged in LDS.
// ---------------------------------------------------------------------------
__global__ __launch_bounds__(256) void gin_agg_kernel(
    const float* __restrict__ x2d, const int* __restrict__ rows,
    const int* __restrict__ cols, const int* __restrict__ eattr,
    const float* __restrict__ bond /* [4][256] */,
    const float* __restrict__ epsArr, int bi, float* __restrict__ hin) {
  const int g = blockIdx.x;
  const int t = (int)threadIdx.x;
  __shared__ float sx[32 * 256];
  __shared__ float sb[4 * 256];
  __shared__ int   sr[EPG], sc[EPG], sa[EPG];

  for (int k = t; k < 32 * 256; k += 256)
    sx[k] = x2d[(size_t)g * 32 * 256 + k];
  for (int k = t; k < 4 * 256; k += 256) sb[k] = bond[k];
  {
    int ge = g * EPG + t;
    sr[t] = rows[ge] - g * 32;
    sc[t] = cols[ge] - g * 32;
    sa[t] = eattr[ge];
  }
  __syncthreads();

  const int r  = t >> 3;          // owned destination row
  const int fb = (t & 7) * 32;    // owned feature slice
  float accv[32];
#pragma unroll
  for (int u = 0; u < 32; ++u) accv[u] = 0.f;

  for (int e = 0; e < EPG; ++e) {
    if (sr[e] == r) {
      const float* xs = &sx[sc[e] * 256 + fb];
      const float* bs = &sb[sa[e] * 256 + fb];
#pragma unroll
      for (int u = 0; u < 32; ++u) accv[u] += fmaxf(xs[u] + bs[u], 0.f);
    }
  }
  const float eps1 = 1.f + epsArr[bi];
  size_t o = ((size_t)(g * 32 + r)) * EMB + fb;
#pragma unroll
  for (int u = 0; u < 32; ++u)
    hin[o + u] = eps1 * sx[r * 256 + fb + u] + accv[u];
}

// ---------------------------------------------------------------------------
// Fused SchNet continuous-filter block, one block per graph:
//   for each target atom i:
//     T1 = ssp(rbf[g,i,:,:] @ w1p + b1)        (32x256, WMMA, K=64)
//     T2 = T1 @ w2 + b2                        (32x256, WMMA, K=256)
//     m[g,i,f] = sum_j T2[j,f] * ccut[g,i,j] * h1[g,j,f]
// ---------------------------------------------------------------------------
__global__ __launch_bounds__(256) void sch_fused_kernel(
    const _Float16* __restrict__ rbf,  // [G*1024][64]
    const float* __restrict__ ccut,    // [G][32][32]
    const float* __restrict__ h1,      // [N][256]
    const _Float16* __restrict__ w1p,  // [64][256]
    const float* __restrict__ b1,
    const _Float16* __restrict__ w2,   // [256][256]
    const float* __restrict__ b2,
    float* __restrict__ mout)          // [N][256]
{
  const int g   = blockIdx.x;
  const int t   = (int)threadIdx.x;
  const int w   = t >> 5;
  const int mt  = w >> 2;
  const int ntb = w & 3;

  __shared__ _Float16 sA[32 * 72];    // rbf tile for current i
  __shared__ _Float16 sT1[32 * 264];  // stage-1 activations (f16)
  __shared__ float    sT2[32 * 264];  // stage-2 filter rows (f32)
  __shared__ _Float16 sH1[32 * 256];  // h1 for this graph
  __shared__ float    sC[1024];       // ccut for this graph

  for (int idx = t; idx < 32 * 256; idx += 256) {
    int rr = idx >> 8, cc = idx & 255;
    sH1[idx] = (_Float16)h1[((size_t)(g * 32 + rr)) * EMB + cc];
  }
  for (int idx = t; idx < 1024; idx += 256)
    sC[idx] = ccut[(size_t)g * 1024 + idx];

  for (int i = 0; i < 32; ++i) {
    __syncthreads();  // previous reduce done before restaging
    {
      int r = t >> 3, c0 = (t & 7) * 8;
      const _Float16* src = &rbf[((size_t)g * 1024 + i * 32 + r) * NGP + c0];
#pragma unroll
      for (int u = 0; u < 8; ++u) sA[r * 72 + c0 + u] = src[u];
    }
    __syncthreads();

    // GEMM1: 32x256, K=64
#pragma unroll
    for (int t4 = 0; t4 < 4; ++t4) {
      int nt = ntb + 4 * t4;
      v8f acc = {};
#pragma unroll
      for (int kc = 0; kc < 64; kc += 32) {
        v16h a = load_frag_a(&sA[(mt * 16) * 72 + kc], 72);
        v16h b = load_frag_b(&w1p[(size_t)kc * EMB + nt * 16], EMB);
        acc = wmma_f16(a, b, acc);
      }
      int lane = t & 31, n = lane & 15, mb = (lane >> 4) * 8;
      int gn = nt * 16 + n;
      float bv = b1[gn];
#pragma unroll
      for (int p = 0; p < 8; ++p)
        sT1[(mt * 16 + mb + p) * 264 + gn] = (_Float16)ssp_f(acc[p] + bv);
    }
    __syncthreads();

    // GEMM2: 32x256, K=256
#pragma unroll
    for (int t4 = 0; t4 < 4; ++t4) {
      int nt = ntb + 4 * t4;
      v8f acc = {};
      for (int kc = 0; kc < 256; kc += 32) {
        v16h a = load_frag_a(&sT1[(mt * 16) * 264 + kc], 264);
        v16h b = load_frag_b(&w2[(size_t)kc * EMB + nt * 16], EMB);
        acc = wmma_f16(a, b, acc);
      }
      int lane = t & 31, n = lane & 15, mb = (lane >> 4) * 8;
      int gn = nt * 16 + n;
      float bv = b2[gn];
#pragma unroll
      for (int p = 0; p < 8; ++p)
        sT2[(mt * 16 + mb + p) * 264 + gn] = acc[p] + bv;
    }
    __syncthreads();

    // weighted neighbor reduction: thread t owns feature f = t
    float s = 0.f;
#pragma unroll
    for (int j = 0; j < 32; ++j)
      s += sT2[j * 264 + t] * sC[i * 32 + j] * (float)sH1[j * 256 + t];
    mout[((size_t)(g * 32 + i)) * EMB + t] = s;
  }
}

// ---------------------------------------------------------------------------
// LayerNorm (+optional residual, +optional relu). Block = D threads.
// ---------------------------------------------------------------------------
template <bool RELU>
__global__ void ln_kernel(const float* __restrict__ In, int ldin,
                          const float* __restrict__ gam,
                          const float* __restrict__ bet,
                          const float* __restrict__ res, int ldres,
                          float* __restrict__ Out, int ldout, int D) {
  const int r = blockIdx.x;
  const int t = (int)threadIdx.x;
  __shared__ float sred[32];
  const int lane = t & 31, w = t >> 5, nw = D >> 5;

  float v = In[(size_t)r * ldin + t];
  float s = v;
#pragma unroll
  for (int o = 16; o; o >>= 1) s += __shfl_down(s, o, 32);
  if (lane == 0) sred[w] = s;
  __syncthreads();
  if (t == 0) {
    float tot = 0.f;
    for (int k = 0; k < nw; ++k) tot += sred[k];
    sred[0] = tot / (float)D;
  }
  __syncthreads();
  const float mu = sred[0];
  __syncthreads();

  float d = v - mu;
  float s2 = d * d;
#pragma unroll
  for (int o = 16; o; o >>= 1) s2 += __shfl_down(s2, o, 32);
  if (lane == 0) sred[w] = s2;
  __syncthreads();
  if (t == 0) {
    float tot = 0.f;
    for (int k = 0; k < nw; ++k) tot += sred[k];
    sred[0] = tot / (float)D;
  }
  __syncthreads();
  const float var = sred[0];

  float o = d * rsqrtf(var + 1e-5f) * gam[t] + bet[t];
  if (res) o += res[(size_t)r * ldres + t];
  if (RELU) o = fmaxf(o, 0.f);
  Out[(size_t)r * ldout + t] = o;
}

// ---------------------------------------------------------------------------
__global__ void cvt_f16_kernel(const float* __restrict__ s,
                               _Float16* __restrict__ d, int n) {
  int i = blockIdx.x * 256 + (int)threadIdx.x;
  if (i < n) d[i] = (_Float16)s[i];
}

__global__ void cvt_pad_kernel(const float* __restrict__ s,
                               _Float16* __restrict__ d, int rows, int cols,
                               int prows) {
  int i = blockIdx.x * 256 + (int)threadIdx.x;
  if (i < prows * cols) {
    int r = i / cols, c = i - r * cols;
    d[i] = (r < rows) ? (_Float16)s[(size_t)r * cols + c] : (_Float16)0.f;
  }
}

__global__ void copy_mat_kernel(const float* __restrict__ src, int lds,
                                float* __restrict__ dst, int ldd, int cols) {
  int r = blockIdx.x;
  for (int c = (int)threadIdx.x; c < cols; c += (int)blockDim.x)
    dst[(size_t)r * ldd + c] = src[(size_t)r * lds + c];
}

__global__ void pool_kernel(const float* __restrict__ x2d,
                            const float* __restrict__ x3f,
                            float* __restrict__ out) {
  const int g = blockIdx.x;
  const int t = (int)threadIdx.x;
  float s2 = 0.f, s3 = 0.f;
  for (int i = 0; i < ATOMS; ++i) {
    size_t o = ((size_t)(g * ATOMS + i)) * EMB + t;
    s2 += x2d[o];
    s3 += x3f[o];
  }
  out[(size_t)g * 512 + t]       = s2 * (1.f / 32.f);
  out[(size_t)g * 512 + EMB + t] = s3 * (1.f / 32.f);
}

}  // namespace

// ===========================================================================
extern "C" void kernel_launch(void* const* d_in, const int* in_sizes, int n_in,
                              void* d_out, int out_size, void* d_ws,
                              size_t ws_size, hipStream_t stream) {
  (void)in_sizes; (void)n_in; (void)out_size; (void)ws_size;

  const int*   xcls  = (const int*)d_in[0];
  const int*   erow  = (const int*)d_in[1];
  const int*   ecol  = erow + E;
  const int*   eatt  = (const int*)d_in[2];
  const float* pos   = (const float*)d_in[3];
  // d_in[4] = batch (atoms contiguous per graph -> not needed)
  const float* P_emb2d   = (const float*)d_in[5];
  const float* P_emb3d   = (const float*)d_in[6];
  const float* P_bond    = (const float*)d_in[7];
  const float* P_eps     = (const float*)d_in[8];
  const float* P_gw1     = (const float*)d_in[9];
  const float* P_gb1     = (const float*)d_in[10];
  const float* P_gw2     = (const float*)d_in[11];
  const float* P_gb2     = (const float*)d_in[12];
  const float* P_sw1     = (const float*)d_in[13];
  const float* P_sb1     = (const float*)d_in[14];
  const float* P_sw2     = (const float*)d_in[15];
  const float* P_sb2     = (const float*)d_in[16];
  const float* P_slin1   = (const float*)d_in[17];
  const float* P_slin2w  = (const float*)d_in[18];
  const float* P_slin2b  = (const float*)d_in[19];
  const float* P_slinw   = (const float*)d_in[20];
  const float* P_slinb   = (const float*)d_in[21];
  const float* P_ln2g    = (const float*)d_in[22];
  const float* P_ln2b    = (const float*)d_in[23];
  const float* P_ln3g    = (const float*)d_in[24];
  const float* P_ln3b    = (const float*)d_in[25];
  const float* P_interw  = (const float*)d_in[26];
  const float* P_interb  = (const float*)d_in[27];
  const float* P_interg  = (const float*)d_in[28];
  const float* P_interbt = (const float*)d_in[29];
  const float* P_l1w     = (const float*)d_in[30];
  const float* P_l1b     = (const float*)d_in[31];
  const float* P_l2w     = (const float*)d_in[32];
  const float* P_l2b     = (const float*)d_in[33];

  // workspace bump allocator
  char* wp = (char*)d_ws;
  auto alloc = [&](size_t bytes) -> void* {
    void* p = (void*)wp;
    wp += (bytes + 255) & ~(size_t)255;
    return p;
  };
  float* x2d   = (float*)alloc((size_t)N * EMB * 4);
  float* x3d   = (float*)alloc((size_t)N * EMB * 4);
  float* prev2 = (float*)alloc((size_t)N * EMB * 4);
  float* prev3 = (float*)alloc((size_t)N * EMB * 4);
  float* hin   = (float*)alloc((size_t)N * EMB * 4);
  float* gbuf  = (float*)alloc((size_t)N * EMB * 4);
  float* h1    = (float*)alloc((size_t)N * EMB * 4);
  float* msch  = (float*)alloc((size_t)N * EMB * 4);
  float* hid   = (float*)alloc((size_t)N * 2 * EMB * 4);
  float* z     = (float*)alloc((size_t)N * 2 * EMB * 4);
  float* ztmp  = (float*)alloc((size_t)N * 2 * EMB * 4);
  float* ccut  = (float*)alloc((size_t)G * 1024 * 4);
  _Float16* rbf   = (_Float16*)alloc((size_t)G * 1024 * NGP * 2);
  _Float16* wbuf  = (_Float16*)alloc((size_t)2 * EMB * 2 * EMB * 2);
  _Float16* wbuf2 = (_Float16*)alloc((size_t)2 * EMB * 2 * EMB * 2);
  _Float16* w1pad = (_Float16*)alloc((size_t)NGP * EMB * 2);

  auto gemm = [&](int act, const float* In, int ldin, const _Float16* Wt,
                  int ldw, const float* bias, float* Out, int ldout, int M,
                  int K, int Nout) {
    dim3 grid(Nout / 64, M / 64);
    if (act == 0)
      gemm_kernel<0><<<grid, 256, 0, stream>>>(In, ldin, Wt, ldw, bias, Out, ldout, K);
    else if (act == 1)
      gemm_kernel<1><<<grid, 256, 0, stream>>>(In, ldin, Wt, ldw, bias, Out, ldout, K);
    else
      gemm_kernel<2><<<grid, 256, 0, stream>>>(In, ldin, Wt, ldw, bias, Out, ldout, K);
  };
  auto cvt = [&](const float* s, _Float16* d, int n) {
    cvt_f16_kernel<<<(n + 255) / 256, 256, 0, stream>>>(s, d, n);
  };

  // ---- geometry + embeddings -------------------------------------------
  geom_kernel<<<G, 256, 0, stream>>>(pos, rbf, ccut);
  embed_kernel<<<N, 256, 0, stream>>>(xcls, P_emb2d, P_emb3d, x2d, x3d, prev2, prev3);

  for (int bi = 0; bi < NB; ++bi) {
    // ---- GIN (2D) ----
    gin_agg_kernel<<<G, 256, 0, stream>>>(x2d, erow, ecol, eatt,
                                          P_bond + (size_t)bi * 4 * EMB,
                                          P_eps, bi, hin);
    cvt(P_gw1 + (size_t)bi * EMB * 2 * EMB, wbuf, EMB * 2 * EMB);
    gemm(1, hin, EMB, wbuf, 2 * EMB, P_gb1 + bi * 2 * EMB, hid, 2 * EMB, N, EMB, 2 * EMB);
    cvt(P_gw2 + (size_t)bi * 2 * EMB * EMB, wbuf2, 2 * EMB * EMB);
    gemm(0, hid, 2 * EMB, wbuf2, EMB, P_gb2 + bi * EMB, gbuf, EMB, N, 2 * EMB, EMB);
    ln_kernel<false><<<N, EMB, 0, stream>>>(gbuf, EMB, P_ln2g + bi * EMB,
                                            P_ln2b + bi * EMB, prev2, EMB,
                                            x2d, EMB, EMB);

    // ---- SchNet (3D) ----
    cvt(P_slin1 + (size_t)bi * EMB * EMB, wbuf2, EMB * EMB);
    gemm(0, x3d, EMB, wbuf2, EMB, nullptr, h1, EMB, N, EMB, EMB);
    cvt_pad_kernel<<<(NGP * EMB + 255) / 256, 256, 0, stream>>>(
        P_sw1 + (size_t)bi * NG * EMB, w1pad, NG, EMB, NGP);
    cvt(P_sw2 + (size_t)bi * EMB * EMB, wbuf, EMB * EMB);
    sch_fused_kernel<<<G, 256, 0, stream>>>(rbf, ccut, h1, w1pad,
                                            P_sb1 + bi * EMB, wbuf,
                                            P_sb2 + bi * EMB, msch);
    cvt(P_slin2w + (size_t)bi * EMB * EMB, wbuf2, EMB * EMB);
    gemm(2, msch, EMB, wbuf2, EMB, P_slin2b + bi * EMB, gbuf, EMB, N, EMB, EMB);
    cvt(P_slinw + (size_t)bi * EMB * EMB, wbuf, EMB * EMB);
    gemm(0, gbuf, EMB, wbuf, EMB, P_slinb + bi * EMB, hin, EMB, N, EMB, EMB);
    ln_kernel<false><<<N, EMB, 0, stream>>>(hin, EMB, P_ln3g + bi * EMB,
                                            P_ln3b + bi * EMB, prev3, EMB,
                                            x3d, EMB, EMB);

    // residual snapshots
    hipMemcpyAsync(prev2, x2d, (size_t)N * EMB * 4, hipMemcpyDeviceToDevice, stream);
    hipMemcpyAsync(prev3, x3d, (size_t)N * EMB * 4, hipMemcpyDeviceToDevice, stream);

    // ---- interactor ----
    copy_mat_kernel<<<N, 256, 0, stream>>>(x2d, EMB, z, 2 * EMB, EMB);
    copy_mat_kernel<<<N, 256, 0, stream>>>(x3d, EMB, z + EMB, 2 * EMB, EMB);
    for (int l = 0; l < 2; ++l) {
      cvt(P_interw + (size_t)l * 2 * EMB * 2 * EMB, wbuf, 2 * EMB * 2 * EMB);
      gemm(0, z, 2 * EMB, wbuf, 2 * EMB, P_interb + l * 2 * EMB, ztmp, 2 * EMB,
           N, 2 * EMB, 2 * EMB);
      ln_kernel<true><<<N, 2 * EMB, 0, stream>>>(ztmp, 2 * EMB,
                                                 P_interg + l * 2 * EMB,
                                                 P_interbt + l * 2 * EMB,
                                                 nullptr, 0, z, 2 * EMB, 2 * EMB);
    }
    copy_mat_kernel<<<N, 256, 0, stream>>>(z, 2 * EMB, x2d, EMB, EMB);
    copy_mat_kernel<<<N, 256, 0, stream>>>(z + EMB, 2 * EMB, x3d, EMB, EMB);
  }

  // ---- final 3D head + pooling -----------------------------------------
  cvt(P_l1w, wbuf, EMB * EMB);
  gemm(2, x3d, EMB, wbuf, EMB, P_l1b, gbuf, EMB, N, EMB, EMB);
  cvt(P_l2w, wbuf2, EMB * EMB);
  gemm(0, gbuf, EMB, wbuf2, EMB, P_l2b, hin, EMB, N, EMB, EMB);
  pool_kernel<<<G, 256, 0, stream>>>(x2d, hin, (float*)d_out);
}
